// TAViT_24893630447881
// MI455X (gfx1250) — matrix operand order repaired and checked by reading, not compiled, not run on hardware
//
#include <hip/hip_runtime.h>

#define PP    64
#define DIM   128
#define ROTD  64
#define RECN  3
#define TT    15
#define HN    256                  // patches per image (16x16)
#define B2    2
#define MROWS (B2*TT*HN)           // 7680
#define PPPK  4096                 // 64*64 patch elements

typedef __attribute__((ext_vector_type(16))) __bf16 v16bf;
typedef __attribute__((ext_vector_type(8)))  float  v8f;

union FragBF { v16bf v; __bf16 e[16]; };

__device__ __forceinline__ __bf16 tobf(float f) { return (__bf16)f; }

// gfx1250 async memory->LDS copy, 16 bytes per lane, tracked by ASYNCcnt.
__device__ __forceinline__ void async_copy_b128(unsigned lds_off, const float* gsrc) {
    asm volatile("global_load_async_to_lds_b128 %0, %1, off"
                 :: "v"(lds_off), "v"((unsigned long long)(size_t)gsrc)
                 : "memory");
}
__device__ __forceinline__ void wait_async0() {
    asm volatile("s_wait_asynccnt 0x0" ::: "memory");
}

// A fragment (16x32 bf16, MxK) built from an f32 LDS tile, contiguous k order (same as B):
// lane L: m = L%16; VGPR j in 0..7 -> K = (L/16)*16 + 2j + {0,1}
__device__ __forceinline__ v16bf a_frag_lds(const float* As, int ldA, int lane, int kb) {
    FragBF a;
    const float* row = As + (lane & 15) * ldA + kb;
    const int hi = (lane >> 4) << 4;                 // 0 or 16
#pragma unroll
    for (int j = 0; j < 4; ++j) {
        a.e[2*j]         = tobf(row[hi + 2*j]);
        a.e[2*j + 1]     = tobf(row[hi + 2*j + 1]);
        a.e[8 + 2*j]     = tobf(row[8 + hi + 2*j]);
        a.e[8 + 2*j + 1] = tobf(row[8 + hi + 2*j + 1]);
    }
    return a.v;
}

// B fragment (32x16 bf16, KxN) from global f32 row-major B (ldB = N):
// lane L: n = L%16; khalf = (L/16)*16; VGPR j -> K = khalf + 2j + {0,1}
__device__ __forceinline__ v16bf b_frag_glb(const float* __restrict__ Bm, int N,
                                            int k0, int n, int lane) {
    FragBF b;
    const int kh = (lane >> 4) << 4;
#pragma unroll
    for (int j = 0; j < 8; ++j) {
        b.e[2*j]     = tobf(Bm[(size_t)(k0 + kh + 2*j)     * N + n]);
        b.e[2*j + 1] = tobf(Bm[(size_t)(k0 + kh + 2*j + 1) * N + n]);
    }
    return b.v;
}

// ---------------------------------------------------------------------------
// Patch embed: C[7680 x 128] = gather(x)[7680 x 4096] @ W_emb + b_emb
// 256 threads / 8 waves per block: 16-row stripe, each wave one 16x16 tile.
// A tile (16 x 64 f32) double-buffered via async global->LDS (b128/lane):
// copy of tile i+1 overlaps the 2x WMMA + B-loads of tile i.
// ---------------------------------------------------------------------------
__global__ __launch_bounds__(256) void embed_kernel(
    const float* __restrict__ x, const float* __restrict__ Wemb,
    const float* __restrict__ bemb, float* __restrict__ xe,
    float* __restrict__ hbuf)
{
    __shared__ float As[2][16][68];                // 68: 16B-aligned rows + pad
    const int m0   = blockIdx.x * 16;
    const int wave = threadIdx.x >> 5;
    const int lane = threadIdx.x & 31;
    const int n    = wave * 16 + (lane & 15);

    // staging assignment: thread -> (row r, 4 consecutive k at kl)
    const int r  = threadIdx.x >> 4;               // 0..15
    const int kl = (threadIdx.x & 15) * 4;         // 0..60
    const unsigned lds_off[2] = { (unsigned)(size_t)&As[0][r][kl],
                                  (unsigned)(size_t)&As[1][r][kl] };

    // row -> patch gather base; k-step of 64 == one 64-wide patch row
    const int row = m0 + r;
    const int bt  = row >> 8;                      // b*15 + t  (c-dim is 1)
    const int p   = row & 255;
    const int ph  = p >> 4, pw = p & 15;
    const float* rbase = x + ((size_t)bt << 20) + (size_t)ph * PP * 1024 + pw * PP + kl;

    async_copy_b128(lds_off[0], rbase);            // prologue: tile 0
    int buf = 0;
    v8f acc = {};
    for (int k0 = 0; k0 < PPPK; k0 += 64) {
        wait_async0();                             // current tile landed (this wave)
        __syncthreads();                           // ... and published block-wide
        if (k0 + 64 < PPPK)                        // issue next tile into other buffer
            async_copy_b128(lds_off[buf ^ 1], rbase + (size_t)((k0 >> 6) + 1) * 1024);
        const float* cur = &As[buf][0][0];
        v16bf a0 = a_frag_lds(cur, 68, lane, 0);
        v16bf b0 = b_frag_glb(Wemb, DIM, k0, n, lane);
        acc = __builtin_amdgcn_wmma_f32_16x16x32_bf16(false, a0, false, b0,
                                                      (short)0, acc, false, false);
        v16bf a1 = a_frag_lds(cur, 68, lane, 32);
        v16bf b1 = b_frag_glb(Wemb, DIM, k0 + 32, n, lane);
        acc = __builtin_amdgcn_wmma_f32_16x16x32_bf16(false, a1, false, b1,
                                                      (short)0, acc, false, false);
        buf ^= 1;
    }
    const float bz = bemb[n];
    const int mbase = m0 + ((lane >> 4) << 3);
#pragma unroll
    for (int j = 0; j < 8; ++j) {
        const float val = acc[j] + bz;
        xe[(size_t)(mbase + j) * DIM + n]   = val;   // pristine x_e output
        hbuf[(size_t)(mbase + j) * DIM + n] = val;   // working recurrence state
    }
}

// ---------------------------------------------------------------------------
// Generic C[M x N] = A[M x K] @ B[K x N] (+ bias), f32 in/out, bf16 WMMA core.
// grid = (N/128, M/16); block = 256 threads; K multiple of 64. Same
// double-buffered async A staging as the embed kernel.
// ---------------------------------------------------------------------------
__global__ __launch_bounds__(256) void gemm_bias_kernel(
    const float* __restrict__ A, const float* __restrict__ Bm,
    const float* __restrict__ bias, float* __restrict__ C,
    int M, int N, int K)
{
    __shared__ float As[2][16][68];
    const int m0   = blockIdx.y * 16;
    const int wave = threadIdx.x >> 5;
    const int lane = threadIdx.x & 31;
    const int n    = blockIdx.x * 128 + wave * 16 + (lane & 15);

    const int r  = threadIdx.x >> 4;
    const int kl = (threadIdx.x & 15) * 4;
    const unsigned lds_off[2] = { (unsigned)(size_t)&As[0][r][kl],
                                  (unsigned)(size_t)&As[1][r][kl] };
    const float* arow = A + (size_t)(m0 + r) * K + kl;

    async_copy_b128(lds_off[0], arow);
    int buf = 0;
    v8f acc = {};
    for (int k0 = 0; k0 < K; k0 += 64) {
        wait_async0();
        __syncthreads();
        if (k0 + 64 < K)
            async_copy_b128(lds_off[buf ^ 1], arow + k0 + 64);
        const float* cur = &As[buf][0][0];
        v16bf a0 = a_frag_lds(cur, 68, lane, 0);
        v16bf b0 = b_frag_glb(Bm, N, k0, n, lane);
        acc = __builtin_amdgcn_wmma_f32_16x16x32_bf16(false, a0, false, b0,
                                                      (short)0, acc, false, false);
        v16bf a1 = a_frag_lds(cur, 68, lane, 32);
        v16bf b1 = b_frag_glb(Bm, N, k0 + 32, n, lane);
        acc = __builtin_amdgcn_wmma_f32_16x16x32_bf16(false, a1, false, b1,
                                                      (short)0, acc, false, false);
        buf ^= 1;
    }
    const float bz = bias ? bias[n] : 0.0f;
    const int mbase = m0 + ((lane >> 4) << 3);
#pragma unroll
    for (int j = 0; j < 8; ++j)
        C[(size_t)(mbase + j) * N + n] = acc[j] + bz;
}

// ---------------------------------------------------------------------------
// q/k/v projections over the 15-channel dim + RoPE on q,k.
// grid = (b*HN) blocks, 128 threads (w). Wave32: RoPE partner = lane^1.
// ---------------------------------------------------------------------------
__global__ __launch_bounds__(128) void qkv_kernel(
    const float* __restrict__ h,
    const float* __restrict__ Wq, const float* __restrict__ bq,
    const float* __restrict__ Wk, const float* __restrict__ bk,
    const float* __restrict__ Wv, const float* __restrict__ bv,
    float* __restrict__ q, float* __restrict__ k, float* __restrict__ v,
    int rec)
{
    const int b  = blockIdx.x >> 8;
    const int hh = blockIdx.x & 255;
    const int w  = threadIdx.x;
    float xin[TT];
#pragma unroll
    for (int c = 0; c < TT; ++c)
        xin[c] = h[(((size_t)b * TT + c) * HN + hh) * DIM + w];

    const float* wq = Wq + rec * TT;
    const float* wk = Wk + rec * TT;
    float qv = bq[rec], kv = bk[rec];
#pragma unroll
    for (int c = 0; c < TT; ++c) { qv += xin[c] * wq[c]; kv += xin[c] * wk[c]; }

    // RoPE on first ROTD dims: rot[2i] = -t[2i+1], rot[2i+1] = t[2i]
    const float qp = __shfl_xor(qv, 1, 32);
    const float kp = __shfl_xor(kv, 1, 32);
    if (w < ROTD) {
        const float inv = __powf(10000.0f, -(float)(w & ~1) / (float)ROTD);
        const float ang = (float)hh * inv;
        const float cs = __cosf(ang), sn = __sinf(ang);
        const float rq = (w & 1) ? qp : -qp;
        const float rk = (w & 1) ? kp : -kp;
        qv = qv * cs + rq * sn;
        kv = kv * cs + rk * sn;
    }
    q[((size_t)b * HN + hh) * DIM + w] = qv;
    k[((size_t)b * HN + hh) * DIM + w] = kv;

    const float* wv  = Wv + rec * TT * TT;
    const float* bvv = bv + rec * TT;
#pragma unroll
    for (int o = 0; o < TT; ++o) {
        float a = bvv[o];
#pragma unroll
        for (int c = 0; c < TT; ++c) a += xin[c] * wv[o * TT + c];
        v[(((size_t)b * TT + o) * HN + hh) * DIM + w] = a;
    }
}

// ---------------------------------------------------------------------------
// Fused criss-cross attention: per (b,hh) block, thread = w.
// logits: eH[g] = q*k[g][w] (+ -inf at g==hh), eW[u] = q*kRow[u];
// 3-pass streaming softmax (max / sum / weighted accumulate), residual+gamma.
// ---------------------------------------------------------------------------
__global__ __launch_bounds__(128) void attn_kernel(
    const float* __restrict__ q, const float* __restrict__ k,
    const float* __restrict__ v, const float* __restrict__ gamma,
    float* __restrict__ h, int rec)
{
    __shared__ float kRow[DIM];
    __shared__ float vRow[TT][DIM];
    const int b  = blockIdx.x >> 8;
    const int hh = blockIdx.x & 255;
    const int w  = threadIdx.x;
    const float* kb = k + (size_t)b * HN * DIM;
    const float* vb = v + (size_t)b * TT * HN * DIM;

    kRow[w] = kb[(size_t)hh * DIM + w];
#pragma unroll
    for (int c = 0; c < TT; ++c)
        vRow[c][w] = vb[((size_t)c * HN + hh) * DIM + w];
    __syncthreads();

    const float qv = q[((size_t)b * HN + hh) * DIM + w];

    float mx = -3.4e38f;
    for (int g = 0; g < HN; ++g)
        if (g != hh) mx = fmaxf(mx, qv * kb[(size_t)g * DIM + w]);
    for (int u = 0; u < DIM; ++u) mx = fmaxf(mx, qv * kRow[u]);

    float Z = 0.0f;
    for (int g = 0; g < HN; ++g)
        if (g != hh) Z += __expf(qv * kb[(size_t)g * DIM + w] - mx);
    for (int u = 0; u < DIM; ++u) Z += __expf(qv * kRow[u] - mx);

    float accH[TT], accW[TT];
#pragma unroll
    for (int c = 0; c < TT; ++c) { accH[c] = 0.0f; accW[c] = 0.0f; }

    for (int g = 0; g < HN; ++g) {
        if (g == hh) continue;
        const float p = __expf(qv * kb[(size_t)g * DIM + w] - mx);
#pragma unroll
        for (int c = 0; c < TT; ++c)
            accH[c] += p * vb[((size_t)c * HN + g) * DIM + w];
    }
    for (int u = 0; u < DIM; ++u) {
        const float p = __expf(qv * kRow[u] - mx);
#pragma unroll
        for (int c = 0; c < TT; ++c) accW[c] += p * vRow[c][u];
    }

    const float gm   = gamma[rec];
    const float invZ = 1.0f / Z;
#pragma unroll
    for (int c = 0; c < TT; ++c) {
        const size_t idx = (((size_t)b * TT + c) * HN + hh) * DIM + w;
        h[idx] = gm * (accH[c] + accW[c]) * invZ + h[idx];
    }
}

// ---------------------------------------------------------------------------
extern "C" void kernel_launch(void* const* d_in, const int* in_sizes, int n_in,
                              void* d_out, int out_size, void* d_ws, size_t ws_size,
                              hipStream_t stream)
{
    (void)in_sizes; (void)n_in; (void)out_size; (void)ws_size;
    const float* x     = (const float*)d_in[0];
    const float* W_emb = (const float*)d_in[1];
    const float* b_emb = (const float*)d_in[2];
    const float* Wq    = (const float*)d_in[3];
    const float* bq    = (const float*)d_in[4];
    const float* Wk    = (const float*)d_in[5];
    const float* bk    = (const float*)d_in[6];
    const float* Wv    = (const float*)d_in[7];
    const float* bv    = (const float*)d_in[8];
    const float* gamma = (const float*)d_in[9];
    const float* W_mlp = (const float*)d_in[10];
    const float* W_p1  = (const float*)d_in[11];
    const float* b_p1  = (const float*)d_in[12];
    const float* W_p2  = (const float*)d_in[13];
    const float* b_p2  = (const float*)d_in[14];

    float* out     = (float*)d_out;
    float* xe_out  = out;                                  // (2,15,256,128)
    float* xh_out  = xe_out + (size_t)MROWS * DIM;         // (2,15,256,128)
    float* xp_out  = xh_out + (size_t)MROWS * DIM;         // rows x 4096 flat
    float* xhp_out = xp_out + (size_t)MROWS * PPPK;

    float* ws   = (float*)d_ws;
    float* hbuf = ws;                                      // MROWS*DIM
    float* qb   = hbuf + (size_t)MROWS * DIM;              // B2*HN*DIM
    float* kb   = qb   + (size_t)B2 * HN * DIM;
    float* vb   = kb   + (size_t)B2 * HN * DIM;            // MROWS*DIM
    float* tmpb = vb   + (size_t)MROWS * DIM;              // MROWS*DIM

    // 1) patch embed (WMMA, K=4096)
    embed_kernel<<<MROWS / 16, 256, 0, stream>>>(x, W_emb, b_emb, xe_out, hbuf);

    // 2) criss-cross recurrence x3
    for (int i = 0; i < RECN; ++i) {
        qkv_kernel<<<B2 * HN, DIM, 0, stream>>>(hbuf, Wq, bq, Wk, bk, Wv, bv,
                                                qb, kb, vb, i);
        attn_kernel<<<B2 * HN, DIM, 0, stream>>>(qb, kb, vb, gamma, hbuf, i);
    }

    // 3) x_h = h @ W_mlp
    gemm_bias_kernel<<<dim3(1, MROWS / 16), 256, 0, stream>>>(
        hbuf, W_mlp, nullptr, xh_out, MROWS, DIM, DIM);

    // 4) x_p = recover(x_e) : (z@W_p1+b1)@W_p2+b2
    gemm_bias_kernel<<<dim3(1, MROWS / 16), 256, 0, stream>>>(
        xe_out, W_p1, b_p1, tmpb, MROWS, DIM, DIM);
    gemm_bias_kernel<<<dim3(PPPK / 128, MROWS / 16), 256, 0, stream>>>(
        tmpb, W_p2, b_p2, xp_out, MROWS, PPPK, DIM);

    // 5) x_h_p = recover(x_h)
    gemm_bias_kernel<<<dim3(1, MROWS / 16), 256, 0, stream>>>(
        xh_out, W_p1, b_p1, tmpb, MROWS, DIM, DIM);
    gemm_bias_kernel<<<dim3(PPPK / 128, MROWS / 16), 256, 0, stream>>>(
        tmpb, W_p2, b_p2, xhp_out, MROWS, PPPK, DIM);
}
